// CognitiveAttention_43834436223780
// MI455X (gfx1250) — matrix operand order repaired and checked by reading, not compiled, not run on hardware
//
#include <hip/hip_runtime.h>

// ---------- types ----------
typedef __attribute__((ext_vector_type(16))) __bf16        v16bf;
typedef __attribute__((ext_vector_type(8)))  float         v8f;
typedef __attribute__((ext_vector_type(4)))  unsigned int  u32x4;

union BF16Tile { v16bf v; u32x4 q[2]; unsigned short us[16]; };

#define HH   768
#define NHH  12
#define HDD  64
#define SS   2048
#define BB   4
#define MROWS (BB * SS)       // 8192

// ---------- helpers ----------
__device__ __forceinline__ unsigned short f2bf(float f) {
    unsigned int u = __builtin_bit_cast(unsigned int, f);
    u += 0x7FFFu + ((u >> 16) & 1u);              // round-to-nearest-even
    return (unsigned short)(u >> 16);
}

__device__ __forceinline__ v8f wmma_bf16(v16bf a, v16bf b, v8f c) {
    return __builtin_amdgcn_wmma_f32_16x16x32_bf16(false, a, false, b, (short)0, c,
                                                   false, false);
}

__device__ __forceinline__ v8f zero8() {
    v8f z;
#pragma unroll
    for (int i = 0; i < 8; ++i) z[i] = 0.0f;
    return z;
}

// CDNA5 async global -> LDS copy (16 bytes per active lane), tracked by ASYNCcnt
__device__ __forceinline__ void async_b128(unsigned lds_off, const unsigned short* g) {
    asm volatile("global_load_async_to_lds_b128 %0, %1, off"
                 :: "v"(lds_off), "v"((unsigned long long)(size_t)g)
                 : "memory");
}
__device__ __forceinline__ void wait_async0() {
    asm volatile("s_wait_asynccnt 0x0" ::: "memory");
}
__device__ __forceinline__ unsigned lds_off32(const void* p) {
    return (unsigned)(size_t)p;      // low 32 bits of generic shared ptr = LDS offset
}

// cross-lane xor shuffle within wave32 via LDS-permute hardware (no LDS memory)
__device__ __forceinline__ float xor_f(float v, int lane, int mask) {
    int src = ((lane ^ mask) << 2);
    int r = __builtin_amdgcn_ds_bpermute(src, __builtin_bit_cast(int, v));
    return __builtin_bit_cast(float, r);
}
__device__ __forceinline__ float rowmax16(float v, int lane) {
#pragma unroll
    for (int m = 1; m < 16; m <<= 1) v = fmaxf(v, xor_f(v, lane, m));
    return v;
}
__device__ __forceinline__ float rowsum16(float v, int lane) {
#pragma unroll
    for (int m = 1; m < 16; m <<= 1) v += xor_f(v, lane, m);
    return v;
}

// A-matrix 16x32 bf16 tile load (row-major src) per ISA lane layout
__device__ __forceinline__ v16bf load_a16x32(const unsigned short* base, int ld,
                                             int m0, int k0, int lane) {
    const unsigned short* p = base + (size_t)(m0 + (lane & 15)) * ld + k0 + ((lane >> 4) << 3);
    BF16Tile t;
    t.q[0] = *(const u32x4*)(p);
    t.q[1] = *(const u32x4*)(p + 16);
    return t.v;
}

// B-matrix 32x16 read from an LDS row-major [rows][stride] staging buffer
__device__ __forceinline__ v16bf lds_b32x16(const unsigned short* row0, int stride,
                                            int nrow, int kcol, int lane) {
    const unsigned short* p = row0 + (size_t)(nrow + (lane & 15)) * stride + kcol + ((lane >> 4) << 3);
    BF16Tile t;
    t.q[0] = *(const u32x4*)(p);
    t.q[1] = *(const u32x4*)(p + 16);
    return t.v;
}

// ---------- kernel 1: fp32 -> bf16 conversions ----------
__global__ void __launch_bounds__(256)
convert_kernel(const float* __restrict__ x,
               const float* __restrict__ Wq, const float* __restrict__ Wk,
               const float* __restrict__ Wv, const float* __restrict__ Wo,
               unsigned short* __restrict__ xb, unsigned short* __restrict__ wqkv,
               unsigned short* __restrict__ wob) {
    const int i = blockIdx.x * 256 + threadIdx.x;
    const int NX = MROWS * HH;
    const int NW = HH * HH;
    if (i < NX) xb[i] = f2bf(x[i]);
    if (i < NW) {
        wqkv[i]          = f2bf(Wq[i]);
        wqkv[NW + i]     = f2bf(Wk[i]);
        wqkv[2 * NW + i] = f2bf(Wv[i]);
        wob[i]           = f2bf(Wo[i]);
    }
}

// ---------- kernel 2: fused QKV projection GEMM (M=8192, N=2304, K=768) ----------
// block = 8 waves, wave tile = 32(M) x 64(N); weight chunk staged per block via
// async->LDS, K-loop unrolled x2 so ping-pong buffers/registers need no copies.
__global__ void __launch_bounds__(256)
qkv_gemm(const unsigned short* __restrict__ xb, const unsigned short* __restrict__ wqkv,
         const float* __restrict__ bq, const float* __restrict__ bk,
         const float* __restrict__ bv,
         unsigned short* __restrict__ qb, unsigned short* __restrict__ kb,
         unsigned short* __restrict__ vT) {
    __shared__ __align__(16) unsigned short bstage[2][64][40];   // 64 N-rows x 32 K

    const int tid  = threadIdx.x;
    const int lane = tid & 31;
    const int w    = tid >> 5;
    const int m0   = (blockIdx.y * 8 + w) * 32;
    const int n0   = blockIdx.x * 64;

    const int srow = tid >> 2;            // 0..63
    const int scol = (tid & 3) * 8;       // halves
    const unsigned short* bsrc = wqkv + (size_t)(n0 + srow) * HH + scol;

    v8f accL[4], accH[4];
#pragma unroll
    for (int t = 0; t < 4; ++t) { accL[t] = zero8(); accH[t] = zero8(); }

    async_b128(lds_off32(&bstage[0][srow][scol]), bsrc);
    wait_async0();
    __syncthreads();

    v16bf aL = load_a16x32(xb, HH, m0,      0, lane);
    v16bf aH = load_a16x32(xb, HH, m0 + 16, 0, lane);

    for (int k0 = 0; k0 < HH; k0 += 64) {
        // ---- half A: consume bstage[0] (k = k0), prefetch k0+32 into bstage[1]
        async_b128(lds_off32(&bstage[1][srow][scol]), bsrc + k0 + 32);
        v16bf aLn = load_a16x32(xb, HH, m0,      k0 + 32, lane);
        v16bf aHn = load_a16x32(xb, HH, m0 + 16, k0 + 32, lane);
#pragma unroll
        for (int t = 0; t < 4; ++t) {
            v16bf bt = lds_b32x16(&bstage[0][0][0], 40, t * 16, 0, lane);
            accL[t] = wmma_bf16(aL, bt, accL[t]);
            accH[t] = wmma_bf16(aH, bt, accH[t]);
        }
        wait_async0();
        __syncthreads();

        // ---- half B: consume bstage[1] (k = k0+32), prefetch k0+64 into bstage[0]
        const bool more = (k0 + 64) < HH;
        if (more) {
            async_b128(lds_off32(&bstage[0][srow][scol]), bsrc + k0 + 64);
            aL = load_a16x32(xb, HH, m0,      k0 + 64, lane);
            aH = load_a16x32(xb, HH, m0 + 16, k0 + 64, lane);
        }
#pragma unroll
        for (int t = 0; t < 4; ++t) {
            v16bf bt = lds_b32x16(&bstage[1][0][0], 40, t * 16, 0, lane);
            accL[t] = wmma_bf16(aLn, bt, accL[t]);
            accH[t] = wmma_bf16(aHn, bt, accH[t]);
        }
        wait_async0();
        __syncthreads();
    }

    const int lhi = lane >> 4, ln = lane & 15;
#pragma unroll
    for (int t = 0; t < 4; ++t) {
        const int n   = n0 + t * 16 + ln;
        const int sel = n / HH;
        const int nn  = n - sel * HH;
        const int h   = nn >> 6, d = nn & 63;
        const float bias = (sel == 0) ? bq[nn] : ((sel == 1) ? bk[nn] : bv[nn]);
#pragma unroll
        for (int r = 0; r < 8; ++r) {
#pragma unroll
            for (int ms = 0; ms < 2; ++ms) {
                const int m = m0 + ms * 16 + r + 8 * lhi;
                const int bi = m >> 11, s = m & (SS - 1);
                const size_t bh = (size_t)(bi * NHH + h);
                const float v = (ms ? accH[t][r] : accL[t][r]) + bias;
                if (sel == 0)      qb[(bh * SS + s) * HDD + d] = f2bf(v * 0.125f);
                else if (sel == 1) kb[(bh * SS + s) * HDD + d] = f2bf(v);
                else               vT[(bh * HDD + d) * SS + s] = f2bf(v);
            }
        }
    }
}

// ---------- kernel 3: flash attention ----------
__global__ void __launch_bounds__(256)
attn_kernel(const unsigned short* __restrict__ qb, const unsigned short* __restrict__ kbm,
            const unsigned short* __restrict__ vT, const int* __restrict__ amask,
            const float* __restrict__ dim_biases, const int* __restrict__ dim_idx,
            unsigned short* __restrict__ ctx) {
    __shared__ __align__(16) unsigned short pbuf[8][16][40];     // per-wave P tile
    __shared__ __align__(16) unsigned short kst[2][32][72];      // 32 kv x 64 hd
    __shared__ __align__(16) unsigned short vst[2][64][40];      // 64 hd x 32 kv

    const int tid  = threadIdx.x;
    const int lane = tid & 31;
    const int w    = tid >> 5;
    const int gid  = blockIdx.x * 8 + w;
    const int qt   = gid & 127;
    const int bh   = gid >> 7;                 // uniform across block
    const int h    = bh % NHH, b = bh / NHH;
    const int q0   = qt * 16;

    const unsigned short* qp = qb  + (size_t)bh * SS * HDD;
    const unsigned short* kp = kbm + (size_t)bh * SS * HDD;
    const unsigned short* vp = vT  + (size_t)bh * HDD * SS;
    const int* am = amask + b * SS;

    const int kr = tid >> 3, kc = (tid & 7) * 8;   // K staging: 32 rows x 64 halves
    const int vr = tid >> 2, vc = (tid & 3) * 8;   // V staging: 64 rows x 32 halves

    float dbias = 0.0f;
    const int di = *dim_idx;
    if (di >= 0 && di < 5) dbias = dim_biases[di * NHH + h];

    const v16bf qa0 = load_a16x32(qp, HDD, q0, 0,  lane);
    const v16bf qa1 = load_a16x32(qp, HDD, q0, 32, lane);

    const int lhi = lane >> 4, ln = lane & 15;

    float rmax[8], rsum[8];
    v8f o[4];
#pragma unroll
    for (int r = 0; r < 8; ++r) { rmax[r] = -3.0e38f; rsum[r] = 0.0f; }
#pragma unroll
    for (int t = 0; t < 4; ++t) o[t] = zero8();

    auto stage = [&](int buf, int kv) {
        async_b128(lds_off32(&kst[buf][kr][kc]), kp + (size_t)(kv + kr) * HDD + kc);
        async_b128(lds_off32(&vst[buf][vr][vc]), vp + (size_t)vr * SS + kv + vc);
    };

    auto process = [&](int buf, int kv) {
        // scores: two 16x16 tiles, K-dim = 64
        v8f c0 = zero8(), c1 = zero8();
        c0 = wmma_bf16(qa0, lds_b32x16(&kst[buf][0][0], 72, 0,  0,  lane), c0);
        c0 = wmma_bf16(qa1, lds_b32x16(&kst[buf][0][0], 72, 0,  32, lane), c0);
        c1 = wmma_bf16(qa0, lds_b32x16(&kst[buf][0][0], 72, 16, 0,  lane), c1);
        c1 = wmma_bf16(qa1, lds_b32x16(&kst[buf][0][0], 72, 16, 32, lane), c1);

        const bool ok0 = am[kv + ln] != 0;
        const bool ok1 = am[kv + 16 + ln] != 0;
#pragma unroll
        for (int r = 0; r < 8; ++r) {
            c0[r] = ok0 ? (c0[r] + dbias) : -3.0e38f;
            c1[r] = ok1 ? (c1[r] + dbias) : -3.0e38f;
        }

        // online softmax
#pragma unroll
        for (int r = 0; r < 8; ++r) {
            float tm = rowmax16(fmaxf(c0[r], c1[r]), lane);
            const float nm   = fmaxf(rmax[r], tm);
            const float p0   = __expf(c0[r] - nm);
            const float p1   = __expf(c1[r] - nm);
            const float corr = __expf(rmax[r] - nm);
            rmax[r] = nm;
            rsum[r] = rsum[r] * corr + rowsum16(p0 + p1, lane);
#pragma unroll
            for (int t = 0; t < 4; ++t) o[t][r] *= corr;
            pbuf[w][r + 8 * lhi][ln]      = f2bf(p0);
            pbuf[w][r + 8 * lhi][16 + ln] = f2bf(p1);
        }
        asm volatile("s_wait_dscnt 0" ::: "memory");

        BF16Tile pa;
        pa.q[0] = *(const u32x4*)(&pbuf[w][ln][lhi * 8]);
        pa.q[1] = *(const u32x4*)(&pbuf[w][ln][lhi * 8 + 16]);

#pragma unroll
        for (int t = 0; t < 4; ++t)
            o[t] = wmma_bf16(pa.v, lds_b32x16(&vst[buf][0][0], 40, t * 16, 0, lane), o[t]);
    };

    stage(0, 0);
    wait_async0();
    __syncthreads();

    for (int kv = 0; kv < SS; kv += 64) {
        // half A: consume buf0 (kv), prefetch kv+32 into buf1 (always in-range)
        stage(1, kv + 32);
        process(0, kv);
        wait_async0();
        __syncthreads();

        // half B: consume buf1 (kv+32), prefetch kv+64 into buf0
        if (kv + 64 < SS) stage(0, kv + 64);
        process(1, kv + 32);
        wait_async0();
        __syncthreads();
    }

    // normalize and scatter ctx [b, s, h*64 + d] as bf16
#pragma unroll
    for (int t = 0; t < 4; ++t) {
#pragma unroll
        for (int r = 0; r < 8; ++r) {
            const int m = q0 + r + 8 * lhi;
            const float val = o[t][r] / rsum[r];
            ctx[((size_t)(b * SS + m)) * HH + h * HDD + t * 16 + ln] = f2bf(val);
        }
    }
}

// ---------- kernel 4: output projection + bias + residual (M=8192,N=768,K=768) ----------
__global__ void __launch_bounds__(256)
out_gemm(const unsigned short* __restrict__ ctx, const unsigned short* __restrict__ wob,
         const float* __restrict__ bo, const float* __restrict__ resid,
         float* __restrict__ y) {
    __shared__ __align__(16) unsigned short bstage[2][64][40];

    const int tid  = threadIdx.x;
    const int lane = tid & 31;
    const int w    = tid >> 5;
    const int m0   = (blockIdx.y * 8 + w) * 32;
    const int n0   = blockIdx.x * 64;

    const int srow = tid >> 2;
    const int scol = (tid & 3) * 8;
    const unsigned short* bsrc = wob + (size_t)(n0 + srow) * HH + scol;

    v8f accL[4], accH[4];
#pragma unroll
    for (int t = 0; t < 4; ++t) { accL[t] = zero8(); accH[t] = zero8(); }

    async_b128(lds_off32(&bstage[0][srow][scol]), bsrc);
    wait_async0();
    __syncthreads();

    v16bf aL = load_a16x32(ctx, HH, m0,      0, lane);
    v16bf aH = load_a16x32(ctx, HH, m0 + 16, 0, lane);

    for (int k0 = 0; k0 < HH; k0 += 64) {
        async_b128(lds_off32(&bstage[1][srow][scol]), bsrc + k0 + 32);
        v16bf aLn = load_a16x32(ctx, HH, m0,      k0 + 32, lane);
        v16bf aHn = load_a16x32(ctx, HH, m0 + 16, k0 + 32, lane);
#pragma unroll
        for (int t = 0; t < 4; ++t) {
            v16bf bt = lds_b32x16(&bstage[0][0][0], 40, t * 16, 0, lane);
            accL[t] = wmma_bf16(aL, bt, accL[t]);
            accH[t] = wmma_bf16(aH, bt, accH[t]);
        }
        wait_async0();
        __syncthreads();

        const bool more = (k0 + 64) < HH;
        if (more) {
            async_b128(lds_off32(&bstage[0][srow][scol]), bsrc + k0 + 64);
            aL = load_a16x32(ctx, HH, m0,      k0 + 64, lane);
            aH = load_a16x32(ctx, HH, m0 + 16, k0 + 64, lane);
        }
#pragma unroll
        for (int t = 0; t < 4; ++t) {
            v16bf bt = lds_b32x16(&bstage[1][0][0], 40, t * 16, 0, lane);
            accL[t] = wmma_bf16(aLn, bt, accL[t]);
            accH[t] = wmma_bf16(aHn, bt, accH[t]);
        }
        wait_async0();
        __syncthreads();
    }

    const int lhi = lane >> 4, ln = lane & 15;
#pragma unroll
    for (int t = 0; t < 4; ++t) {
        const int n = n0 + t * 16 + ln;
        const float bias = bo[n];
#pragma unroll
        for (int r = 0; r < 8; ++r) {
#pragma unroll
            for (int ms = 0; ms < 2; ++ms) {
                const int m = m0 + ms * 16 + r + 8 * lhi;
                const float v = (ms ? accH[t][r] : accL[t][r]);
                y[(size_t)m * HH + n] = v + bias + resid[(size_t)m * HH + n];
            }
        }
    }
}

// ---------- kernel 5: layernorm over H=768 (one block per row) ----------
__global__ void __launch_bounds__(256)
ln_kernel(const float* __restrict__ y, const float* __restrict__ g,
          const float* __restrict__ be, float* __restrict__ out) {
    __shared__ float red[256];
    const int row = blockIdx.x;
    const int t   = threadIdx.x;
    const float* yr = y + (size_t)row * HH;

    const float a0 = yr[t], a1 = yr[t + 256], a2 = yr[t + 512];
    red[t] = a0 + a1 + a2;
    __syncthreads();
    for (int o = 128; o > 0; o >>= 1) { if (t < o) red[t] += red[t + o]; __syncthreads(); }
    const float mu = red[0] * (1.0f / HH);
    __syncthreads();

    const float d0 = a0 - mu, d1 = a1 - mu, d2 = a2 - mu;
    red[t] = d0 * d0 + d1 * d1 + d2 * d2;
    __syncthreads();
    for (int o = 128; o > 0; o >>= 1) { if (t < o) red[t] += red[t + o]; __syncthreads(); }
    const float inv = rsqrtf(red[0] * (1.0f / HH) + 1e-5f);

    float* orow = out + (size_t)row * HH;
    orow[t]       = d0 * inv * g[t]       + be[t];
    orow[t + 256] = d1 * inv * g[t + 256] + be[t + 256];
    orow[t + 512] = d2 * inv * g[t + 512] + be[t + 512];
}

// ---------- launcher ----------
extern "C" void kernel_launch(void* const* d_in, const int* in_sizes, int n_in,
                              void* d_out, int out_size, void* d_ws, size_t ws_size,
                              hipStream_t stream) {
    (void)in_sizes; (void)n_in; (void)out_size; (void)ws_size;
    const float* x    = (const float*)d_in[0];
    const float* Wq   = (const float*)d_in[1];
    const float* bq   = (const float*)d_in[2];
    const float* Wk   = (const float*)d_in[3];
    const float* bk   = (const float*)d_in[4];
    const float* Wv   = (const float*)d_in[5];
    const float* bv   = (const float*)d_in[6];
    const float* Wo   = (const float*)d_in[7];
    const float* bo   = (const float*)d_in[8];
    const float* dimb = (const float*)d_in[9];
    const float* g    = (const float*)d_in[10];
    const float* be   = (const float*)d_in[11];
    const int*   am   = (const int*)d_in[12];
    const int*   didx = (const int*)d_in[13];
    float* out = (float*)d_out;

    char* ws = (char*)d_ws;
    size_t off = 0;
    auto alloc = [&](size_t bytes) {
        char* p = ws + off;
        off = (off + bytes + 255) & ~(size_t)255;
        return p;
    };
    unsigned short* xb   = (unsigned short*)alloc((size_t)MROWS * HH * 2);
    unsigned short* wqkv = (unsigned short*)alloc((size_t)3 * HH * HH * 2);
    unsigned short* wob  = (unsigned short*)alloc((size_t)HH * HH * 2);
    unsigned short* qbuf = (unsigned short*)alloc((size_t)BB * NHH * SS * HDD * 2);
    unsigned short* kbuf = (unsigned short*)alloc((size_t)BB * NHH * SS * HDD * 2);
    unsigned short* vTb  = (unsigned short*)alloc((size_t)BB * NHH * HDD * SS * 2);
    unsigned short* ctx  = (unsigned short*)alloc((size_t)MROWS * HH * 2);
    float*          ybuf = (float*)alloc((size_t)MROWS * HH * 4);

    convert_kernel<<<(MROWS * HH) / 256, 256, 0, stream>>>(x, Wq, Wk, Wv, Wo, xb, wqkv, wob);
    qkv_gemm<<<dim3(36, 32), 256, 0, stream>>>(xb, wqkv, bq, bk, bv, qbuf, kbuf, vTb);
    attn_kernel<<<768, 256, 0, stream>>>(qbuf, kbuf, vTb, am, dimb, didx, ctx);
    out_gemm<<<dim3(12, 32), 256, 0, stream>>>(ctx, wob, bo, x, ybuf);
    ln_kernel<<<MROWS, 256, 0, stream>>>(ybuf, g, be, out);
}
